// BaichuanLayer_50379966382412
// MI455X (gfx1250) — compile-verified, matrix operand order checked by reading, not run on hardware
//
#include <hip/hip_runtime.h>
#include <stdint.h>

// ---------------- types ----------------
typedef __attribute__((ext_vector_type(16))) __bf16 v16bf;
typedef __attribute__((ext_vector_type(8)))  __bf16 v8bf;
typedef __attribute__((ext_vector_type(8)))  float  v8f;

#define BM 128
#define BN 128
#define BK 32

__device__ __forceinline__ unsigned short f2bf(float f) {
    unsigned int u = __float_as_uint(f);
    u += 0x7FFFu + ((u >> 16) & 1u);          // round-to-nearest-even
    return (unsigned short)(u >> 16);
}
__device__ __forceinline__ float bf2f(unsigned short h) {
    return __uint_as_float(((unsigned int)h) << 16);
}
// packed f32x2 -> bf16x2: RNE-round both, then one v_perm_b32 grabs the two
// high halves ({hi[31:16], lo[31:16]}): dst bytes = {b.b3,b.b2,a.b3,a.b2}
__device__ __forceinline__ unsigned int f2bf_pk(float lo, float hi) {
    unsigned int a = __float_as_uint(lo);
    unsigned int b = __float_as_uint(hi);
    a += 0x7FFFu + ((a >> 16) & 1u);
    b += 0x7FFFu + ((b >> 16) & 1u);
#if __has_builtin(__builtin_amdgcn_perm)
    return __builtin_amdgcn_perm(b, a, 0x07060302u);   // one v_perm_b32
#else
    return (b & 0xFFFF0000u) | (a >> 16);
#endif
}

// Load one 16x32 bf16 WMMA operand fragment from LDS (row-major, stride BK).
// Lanes 0-15 -> rows 0-15, K = {0..7, 16..23}; lanes 16-31 -> rows, K = {8..15, 24..31}.
__device__ __forceinline__ v16bf load_frag(const unsigned short* lds, int row0, int lane) {
    int r  = row0 + (lane & 15);
    int kb = (lane & 16) ? 8 : 0;
    const unsigned short* p = lds + r * BK + kb;
    v16bf f;
    ((v8bf*)&f)[0] = *(const v8bf*)(p);
    ((v8bf*)&f)[1] = *(const v8bf*)(p + 16);
    return f;
}

enum { B_F32_N = 0, B_BF16_N = 1, B_BF16_K = 2 };
enum { EPI_BF16 = 0, EPI_SCOREMASK = 1, EPI_F32RES = 2 };

// Generic WMMA GEMM: C[M,N] = epi( A_bf16[M,K] @ B[K,N] )
// 128x128 block tile, 8 wave32 waves, each wave 32x64 (2x4 16x16 tiles).
// Double-buffered LDS with register-staged prefetch of tile kt+1.
template <int BKIND, int EPI>
__global__ __launch_bounds__(256, 1)
void gemm_wmma(const unsigned short* __restrict__ A, long lda, long a_hs,
               const void* __restrict__ Bv, long ldb, long b_hs,
               void* __restrict__ Cv, long ldc, long c_hs,
               const float* __restrict__ mask, long mask_ld, long m_hs,
               const float* __restrict__ resid,
               float scale, int K)
{
    __shared__ __align__(16) unsigned short sA [2][BM * BK];
    __shared__ __align__(16) unsigned short sBT[2][BN * BK];

    const int tid  = threadIdx.x;
    const int lane = tid & 31;
    const int wave = tid >> 5;       // 0..7
    const int wm   = wave & 3;       // row group: wm*32
    const int wn   = wave >> 2;      // col group: wn*64

    const int z  = blockIdx.z;
    const unsigned short* Ah = A + (size_t)z * a_hs;
    const int m0 = blockIdx.y * BM;
    const int n0 = blockIdx.x * BN;

    // staging register layout
    const int ar = tid >> 1, ac = (tid & 1) * 16;          // A: 16 bf16 strip
    const int bl = tid & 127, bkc = (tid >> 7) * 16;       // B k-strip: col bl, k bkc..bkc+15
    const int br = tid >> 1, bc = (tid & 1) * 16;          // B N-major direct copy

    uint4 ra0, ra1;
    float rbf[16];            // B_F32_N
    uint4 rb0, rb1;           // B_BF16_N
    unsigned short rbs[16];   // B_BF16_K

    v8f acc[2][4] = {};
    const int nK = K / BK;

    auto fetch = [&](int kt) {
        const int k0 = kt * BK;
        const uint4* gp = (const uint4*)(Ah + (size_t)(m0 + ar) * lda + k0 + ac);
        ra0 = gp[0]; ra1 = gp[1];
        if (BKIND == B_F32_N) {
            const float* Bh = (const float*)Bv + (size_t)z * b_hs;
            const float* bp = Bh + (size_t)(k0 + bkc) * ldb + n0 + bl;
            #pragma unroll
            for (int j = 0; j < 16; ++j) rbf[j] = bp[(size_t)j * ldb];
        } else if (BKIND == B_BF16_N) {
            const unsigned short* Bh = (const unsigned short*)Bv + (size_t)z * b_hs;
            const uint4* bp = (const uint4*)(Bh + (size_t)(n0 + br) * ldb + k0 + bc);
            rb0 = bp[0]; rb1 = bp[1];
        } else {
            const unsigned short* Bh = (const unsigned short*)Bv + (size_t)z * b_hs;
            const unsigned short* bp = Bh + (size_t)(k0 + bkc) * ldb + n0 + bl;
            #pragma unroll
            for (int j = 0; j < 16; ++j) rbs[j] = bp[(size_t)j * ldb];
        }
    };
    auto commit = [&](int buf) {
        *(uint4*)&sA[buf][ar * BK + ac]     = ra0;
        *(uint4*)&sA[buf][ar * BK + ac + 8] = ra1;
        if (BKIND == B_F32_N) {
            unsigned int w[8];
            #pragma unroll
            for (int j = 0; j < 8; ++j) w[j] = f2bf_pk(rbf[2 * j], rbf[2 * j + 1]);
            *(uint4*)&sBT[buf][bl * BK + bkc]     = *(uint4*)&w[0];
            *(uint4*)&sBT[buf][bl * BK + bkc + 8] = *(uint4*)&w[4];
        } else if (BKIND == B_BF16_N) {
            *(uint4*)&sBT[buf][br * BK + bc]     = rb0;
            *(uint4*)&sBT[buf][br * BK + bc + 8] = rb1;
        } else {
            *(uint4*)&sBT[buf][bl * BK + bkc]     = *(uint4*)&rbs[0];
            *(uint4*)&sBT[buf][bl * BK + bkc + 8] = *(uint4*)&rbs[8];
        }
    };

    fetch(0);
    commit(0);
    for (int kt = 0; kt < nK; ++kt) {
        __syncthreads();
        const int cur = kt & 1;
        const bool more = (kt + 1 < nK);
        if (more) fetch(kt + 1);            // global loads overlap with compute below

        v16bf af[2], bfr[4];
        #pragma unroll
        for (int i = 0; i < 2; ++i) af[i]  = load_frag(&sA [cur][0], wm * 32 + i * 16, lane);
        #pragma unroll
        for (int j = 0; j < 4; ++j) bfr[j] = load_frag(&sBT[cur][0], wn * 64 + j * 16, lane);
        #pragma unroll
        for (int i = 0; i < 2; ++i)
            #pragma unroll
            for (int j = 0; j < 4; ++j)
                acc[i][j] = __builtin_amdgcn_wmma_f32_16x16x32_bf16(
                    false, af[i], false, bfr[j], (short)0, acc[i][j], false, false);

        if (more) commit(cur ^ 1);
    }

    // ---- epilogue (C layout: lane&15 = N col; VGPR r = M row, +8 for lanes>=16) ----
    const int l15 = lane & 15;
    const int rhi = (lane >> 4) * 8;
    #pragma unroll
    for (int i = 0; i < 2; ++i) {
        #pragma unroll
        for (int j = 0; j < 4; ++j) {
            #pragma unroll
            for (int r = 0; r < 8; ++r) {
                int row = m0 + wm * 32 + i * 16 + r + rhi;
                int col = n0 + wn * 64 + j * 16 + l15;
                float v = acc[i][j][r];
                if (EPI == EPI_BF16) {
                    unsigned short* C = (unsigned short*)Cv + (size_t)z * c_hs;
                    C[(size_t)row * ldc + col] = f2bf(v);
                } else if (EPI == EPI_SCOREMASK) {
                    const float* mh = mask + (size_t)z * m_hs;
                    float mv = mh[(size_t)row * mask_ld + col];
                    unsigned short* C = (unsigned short*)Cv + (size_t)z * c_hs;
                    C[(size_t)row * ldc + col] = f2bf(v * scale + mv);
                } else { // EPI_F32RES
                    float* C = (float*)Cv + (size_t)z * c_hs;
                    size_t idx = (size_t)row * ldc + col;
                    C[idx] = resid[idx] + v;
                }
            }
        }
    }
}

// Fused gate/up GEMM: C = bf16( silu(A@Bg) * (A@Bu) ), double buffered
__global__ __launch_bounds__(256, 1)
void gemm_gateup(const unsigned short* __restrict__ A, long lda,
                 const float* __restrict__ Bg, const float* __restrict__ Bu, long ldb,
                 unsigned short* __restrict__ C, long ldc, int K)
{
    __shared__ __align__(16) unsigned short sA[2][BM * BK];
    __shared__ __align__(16) unsigned short sG[2][BN * BK];
    __shared__ __align__(16) unsigned short sU[2][BN * BK];

    const int tid  = threadIdx.x;
    const int lane = tid & 31;
    const int wave = tid >> 5;
    const int wm   = wave & 3;
    const int wn   = wave >> 2;
    const int m0 = blockIdx.y * BM;
    const int n0 = blockIdx.x * BN;

    const int ar = tid >> 1, ac = (tid & 1) * 16;
    const int bl = tid & 127, bkc = (tid >> 7) * 16;

    uint4 ra0, ra1;
    float rg[16], ru[16];

    v8f ag[2][4] = {};
    v8f au[2][4] = {};
    const int nK = K / BK;

    auto fetch = [&](int kt) {
        const int k0 = kt * BK;
        const uint4* gp4 = (const uint4*)(A + (size_t)(m0 + ar) * lda + k0 + ac);
        ra0 = gp4[0]; ra1 = gp4[1];
        const float* gp = Bg + (size_t)(k0 + bkc) * ldb + n0 + bl;
        const float* up = Bu + (size_t)(k0 + bkc) * ldb + n0 + bl;
        #pragma unroll
        for (int j = 0; j < 16; ++j) { rg[j] = gp[(size_t)j * ldb]; ru[j] = up[(size_t)j * ldb]; }
    };
    auto commit = [&](int buf) {
        *(uint4*)&sA[buf][ar * BK + ac]     = ra0;
        *(uint4*)&sA[buf][ar * BK + ac + 8] = ra1;
        unsigned int wg[8], wu[8];
        #pragma unroll
        for (int j = 0; j < 8; ++j) { wg[j] = f2bf_pk(rg[2*j], rg[2*j+1]);
                                      wu[j] = f2bf_pk(ru[2*j], ru[2*j+1]); }
        *(uint4*)&sG[buf][bl * BK + bkc]     = *(uint4*)&wg[0];
        *(uint4*)&sG[buf][bl * BK + bkc + 8] = *(uint4*)&wg[4];
        *(uint4*)&sU[buf][bl * BK + bkc]     = *(uint4*)&wu[0];
        *(uint4*)&sU[buf][bl * BK + bkc + 8] = *(uint4*)&wu[4];
    };

    fetch(0);
    commit(0);
    for (int kt = 0; kt < nK; ++kt) {
        __syncthreads();
        const int cur = kt & 1;
        const bool more = (kt + 1 < nK);
        if (more) fetch(kt + 1);

        v16bf af[2], gf[4], uf[4];
        #pragma unroll
        for (int i = 0; i < 2; ++i) af[i] = load_frag(&sA[cur][0], wm * 32 + i * 16, lane);
        #pragma unroll
        for (int j = 0; j < 4; ++j) { gf[j] = load_frag(&sG[cur][0], wn * 64 + j * 16, lane);
                                      uf[j] = load_frag(&sU[cur][0], wn * 64 + j * 16, lane); }
        #pragma unroll
        for (int i = 0; i < 2; ++i)
            #pragma unroll
            for (int j = 0; j < 4; ++j) {
                ag[i][j] = __builtin_amdgcn_wmma_f32_16x16x32_bf16(
                    false, af[i], false, gf[j], (short)0, ag[i][j], false, false);
                au[i][j] = __builtin_amdgcn_wmma_f32_16x16x32_bf16(
                    false, af[i], false, uf[j], (short)0, au[i][j], false, false);
            }

        if (more) commit(cur ^ 1);
    }

    const int l15 = lane & 15;
    const int rhi = (lane >> 4) * 8;
    #pragma unroll
    for (int i = 0; i < 2; ++i)
        #pragma unroll
        for (int j = 0; j < 4; ++j)
            #pragma unroll
            for (int r = 0; r < 8; ++r) {
                int row = m0 + wm * 32 + i * 16 + r + rhi;
                int col = n0 + wn * 64 + j * 16 + l15;
                float g = ag[i][j][r];
                float u = au[i][j][r];
                float s = g * __builtin_amdgcn_rcpf(1.0f + __expf(-g));  // silu, v_rcp_f32
                C[(size_t)row * ldc + col] = f2bf(s * u);
            }
}

// RMSNorm: one block per row of n f32 -> bf16 out
__global__ __launch_bounds__(256)
void rmsnorm_kernel(const float* __restrict__ x, const float* __restrict__ w,
                    unsigned short* __restrict__ y, int n)
{
    __shared__ float red[256];
    const float* xr = x + (size_t)blockIdx.x * n;
    unsigned short* yr = y + (size_t)blockIdx.x * n;
    int t = threadIdx.x;
    float s = 0.0f;
    for (int i = t; i < n; i += 256) { float v = xr[i]; s += v * v; }
    red[t] = s; __syncthreads();
    for (int k = 128; k > 0; k >>= 1) { if (t < k) red[t] += red[t + k]; __syncthreads(); }
    float inv = rsqrtf(red[0] / (float)n + 1e-6f);
    for (int i = t; i < n; i += 256) yr[i] = f2bf(w[i] * xr[i] * inv);
}

// In-place row softmax over n bf16 values (scores -> probabilities)
__global__ __launch_bounds__(256)
void softmax_rows(unsigned short* __restrict__ S, int n)
{
    __shared__ float red[256];
    unsigned short* row = S + (size_t)blockIdx.x * n;
    int t = threadIdx.x;
    float vals[4];
    float m = -3.4e38f;
    #pragma unroll
    for (int j = 0; j < 4; ++j) { vals[j] = bf2f(row[t + j * 256]); m = fmaxf(m, vals[j]); }
    red[t] = m; __syncthreads();
    for (int k = 128; k > 0; k >>= 1) { if (t < k) red[t] = fmaxf(red[t], red[t + k]); __syncthreads(); }
    m = red[0]; __syncthreads();
    float sum = 0.0f;
    #pragma unroll
    for (int j = 0; j < 4; ++j) { vals[j] = __expf(vals[j] - m); sum += vals[j]; }
    red[t] = sum; __syncthreads();
    for (int k = 128; k > 0; k >>= 1) { if (t < k) red[t] += red[t + k]; __syncthreads(); }
    float inv = __builtin_amdgcn_rcpf(red[0]);
    #pragma unroll
    for (int j = 0; j < 4; ++j) row[t + j * 256] = f2bf(vals[j] * inv);
}

// ---------------- host launcher ----------------
extern "C" void kernel_launch(void* const* d_in, const int* in_sizes, int n_in,
                              void* d_out, int out_size, void* d_ws, size_t ws_size,
                              hipStream_t stream)
{
    const float* hs     = (const float*)d_in[0];   // [1,1024,5120]
    const float* mask   = (const float*)d_in[1];   // [1,40,1024,1024]
    const float* w_pack = (const float*)d_in[2];   // [5120,15360]
    const float* o_proj = (const float*)d_in[3];   // [5120,5120]
    const float* gate   = (const float*)d_in[4];   // [5120,13696]
    const float* up     = (const float*)d_in[5];   // [5120,13696]
    const float* down   = (const float*)d_in[6];   // [13696,5120]
    const float* ln1    = (const float*)d_in[7];
    const float* ln2    = (const float*)d_in[8];
    float* out = (float*)d_out;

    char* ws = (char*)d_ws;
    const size_t MB = 1ull << 20;
    unsigned short* X   = (unsigned short*)(ws + 0);        // 10.5 MB (later Y)
    unsigned short* QKV = (unsigned short*)(ws + 11 * MB);  // 31.5 MB (later GU)
    unsigned short* SC  = (unsigned short*)(ws + 43 * MB);  // 84 MB scores/P
    unsigned short* AO  = (unsigned short*)(ws + 128 * MB); // 10.5 MB
    float*          H2  = (float*)        (ws + 139 * MB);  // 21 MB
    unsigned short* GU  = QKV;
    unsigned short* Y   = X;

    dim3 blk(256);
    const long HS2 = 1024L * 1024L;

    // 1) x = rmsnorm(hidden, ln1) -> bf16
    rmsnorm_kernel<<<1024, blk, 0, stream>>>(hs, ln1, X, 5120);

    // 2) QKV = X @ w_pack            (M=1024, N=15360, K=5120)
    gemm_wmma<B_F32_N, EPI_BF16><<<dim3(120, 8, 1), blk, 0, stream>>>(
        X, 5120, 0, w_pack, 15360, 0, QKV, 15360, 0,
        nullptr, 0, 0, nullptr, 1.0f, 5120);

    // 3) scores[h] = Q[h] @ K[h]^T * scale + mask[h]   (per-head M=N=1024, K=128)
    gemm_wmma<B_BF16_N, EPI_SCOREMASK><<<dim3(8, 8, 40), blk, 0, stream>>>(
        QKV, 15360, 128, QKV + 5120, 15360, 128, SC, 1024, HS2,
        mask, 1024, HS2, nullptr, 0.08838834764831845f, 128);

    // 4) softmax rows in place
    softmax_rows<<<40 * 1024, blk, 0, stream>>>(SC, 1024);

    // 5) AO[h] = P[h] @ V[h]          (per-head M=1024, N=128, K=1024)
    gemm_wmma<B_BF16_K, EPI_BF16><<<dim3(1, 8, 40), blk, 0, stream>>>(
        SC, 1024, HS2, QKV + 2 * 5120, 15360, 128, AO, 5120, 128,
        nullptr, 0, 0, nullptr, 1.0f, 1024);

    // 6) H2 = hidden + AO @ o_proj    (M=1024, N=5120, K=5120)
    gemm_wmma<B_F32_N, EPI_F32RES><<<dim3(40, 8, 1), blk, 0, stream>>>(
        AO, 5120, 0, o_proj, 5120, 0, H2, 5120, 0,
        nullptr, 0, 0, hs, 1.0f, 5120);

    // 7) y = rmsnorm(H2, ln2) -> bf16
    rmsnorm_kernel<<<1024, blk, 0, stream>>>(H2, ln2, Y, 5120);

    // 8) GU = silu(Y@gate) * (Y@up)   (M=1024, N=13696, K=5120)
    gemm_gateup<<<dim3(107, 8, 1), blk, 0, stream>>>(
        Y, 5120, gate, up, 13696, GU, 13696, 5120);

    // 9) out = H2 + GU @ down         (M=1024, N=5120, K=13696)
    gemm_wmma<B_F32_N, EPI_F32RES><<<dim3(40, 8, 1), blk, 0, stream>>>(
        GU, 13696, 0, down, 5120, 0, out, 5120, 0,
        nullptr, 0, 0, H2, 1.0f, 13696);
}
